// MultiHeadAttention_82059645157766
// MI455X (gfx1250) — matrix operand order repaired and checked by reading, not compiled, not run on hardware
//
#include <hip/hip_runtime.h>
#include <math.h>

// ---------------------------------------------------------------------------
// MI455X (gfx1250) fused multi-head attention, bf16 WMMA + f32 accumulation.
//   B=2, S=2048, D=1024, H=16, hd=64.
// Matmuls: v_wmma_f32_16x16x32_bf16 (wave32).  Tile staging: Tensor Data
// Mover (tensor_load_to_lds / TENSORcnt) with double buffering.
// ---------------------------------------------------------------------------

#define EMBED   1024
#define NHEADS  16
#define HDIM    64
#define SEQ     2048
#define BATCH   2

typedef unsigned short bhalf;   // bf16 bit pattern, kept as raw u16
typedef __attribute__((ext_vector_type(16))) __bf16 v16bf;
typedef __attribute__((ext_vector_type(8)))  float  v8f;
typedef __attribute__((ext_vector_type(4)))  unsigned int u32x4;
typedef __attribute__((ext_vector_type(8)))  int i32x8;
typedef __attribute__((ext_vector_type(4)))  int i32x4;

#if defined(__has_builtin)
#  if __has_builtin(__builtin_amdgcn_tensor_load_to_lds) && \
      __has_builtin(__builtin_amdgcn_s_wait_tensorcnt)
#    define HAVE_TDM 1
#  else
#    define HAVE_TDM 0
#  endif
#else
#  define HAVE_TDM 0
#endif

union FragU { v16bf v; unsigned int u[8]; };

// float -> bf16: native v_cvt on device, integer RNE fallback on host pass
static __device__ __forceinline__ bhalf f2bf(float f) {
#if defined(__AMDGCN__)
    union { __bf16 h; bhalf u; } c;
    c.h = (__bf16)f;
    return c.u;
#else
    union { float f; unsigned int u; } c; c.f = f;
    unsigned int u = c.u;
    return (bhalf)((u + 0x7FFFu + ((u >> 16) & 1u)) >> 16);
#endif
}

// ---------------------------------------------------------------------------
// WMMA fragment gathers from LDS (bf16 as u16, read as u32 pairs; contiguous
// runs merge into ds_load_b128).
// A 16x32 bf16: lane L holds row M=L&15;
//   VGPR j (j<4): K = 2j + 8*(L>>4);  (j>=4): K = 16 + 2(j-4) + 8*(L>>4).
// B 32x16 bf16 (from K-contiguous tile): lane L holds col N=L&15;
//   VGPR j (j<4): K = 2j + 8*(L>>4);  (j>=4): K = 16 + 2(j-4) + 8*(L>>4), element at lds[c*ld + kk].
// C/D 16x16 f32: lane L, VGPR j -> row M = j + 8*(L>>4), col N = L&15.
// ---------------------------------------------------------------------------
static __device__ __forceinline__ v16bf lds_frag_a(const bhalf* lds, int ld,
                                                   int row0, int k0, int lane) {
    const int r  = row0 + (lane & 15);
    const int kh = (lane >> 4) << 3;
    FragU f;
#pragma unroll
    for (int j = 0; j < 8; ++j) {
        const int kk = k0 + kh + ((j < 4) ? (j << 1) : (16 + ((j - 4) << 1)));
        f.u[j] = *(const unsigned int*)(lds + r * ld + kk);
    }
    return f.v;
}

static __device__ __forceinline__ v16bf lds_frag_bT(const bhalf* lds, int ld,
                                                    int k0, int col0, int lane) {
    const int c  = col0 + (lane & 15);
    const int kh = (lane >> 4) << 3;
    FragU f;
#pragma unroll
    for (int j = 0; j < 8; ++j) {
        const int kk = k0 + kh + ((j < 4) ? (j << 1) : (16 + ((j - 4) << 1)));
        f.u[j] = *(const unsigned int*)(lds + c * ld + kk);
    }
    return f.v;
}

static __device__ __forceinline__ v8f wmma_bf16(v16bf a, v16bf b, v8f c) {
    return __builtin_amdgcn_wmma_f32_16x16x32_bf16(false, a, false, b,
                                                   (short)0, c, false, false);
}

// ---------------------------------------------------------------------------
// Tile staging: global -> LDS (VALU fallback paths).
// ---------------------------------------------------------------------------
static __device__ __forceinline__ void stage_tile(const float* g, int ldg,
                                                  bhalf* lds, int ldl,
                                                  int rows, int cols,
                                                  int tid, int nthr) {
    const int total = rows * cols;
    for (int i = tid * 4; i < total; i += nthr * 4) {
        const int r = i / cols, c = i % cols;
        const float4 v = *(const float4*)(g + r * ldg + c);
        bhalf* d = lds + r * ldl + c;
        d[0] = f2bf(v.x); d[1] = f2bf(v.y); d[2] = f2bf(v.z); d[3] = f2bf(v.w);
    }
}

// f32 [K][N] tile -> bf16 LDS stored transposed [N][K] (K-contiguous)
static __device__ __forceinline__ void stage_w_T(const float* g, int ldg,
                                                 bhalf* lds, int ldl,
                                                 int rows, int cols,
                                                 int tid, int nthr) {
    const int total = rows * cols;
    for (int i = tid * 4; i < total; i += nthr * 4) {
        const int r = i / cols, c = i % cols;
        const float4 v = *(const float4*)(g + r * ldg + c);
        lds[(c + 0) * ldl + r] = f2bf(v.x);
        lds[(c + 1) * ldl + r] = f2bf(v.y);
        lds[(c + 2) * ldl + r] = f2bf(v.z);
        lds[(c + 3) * ldl + r] = f2bf(v.w);
    }
}

static __device__ __forceinline__ void stage_tile(const bhalf* g, int ldg,
                                                  bhalf* lds, int ldl,
                                                  int rows, int cols,
                                                  int tid, int nthr) {
    const int total = rows * cols;
    for (int i = tid * 8; i < total; i += nthr * 8) {
        const int r = i / cols, c = i % cols;
        *(uint4*)(lds + r * ldl + c) = *(const uint4*)(g + r * ldg + c);
    }
}

static __device__ __forceinline__ void store_out(float* p, float v) { *p = v; }
static __device__ __forceinline__ void store_out(bhalf* p, float v) { *p = f2bf(v); }

// ---------------------------------------------------------------------------
// Tensor Data Mover: 2D bf16 tile load, global -> LDS, with LDS row padding
// (pad_amount code 3 = 4 DWORDs = 8 bf16 elems appended per row):
//   64-elem rows (pi_code=4): padded LDS stride 72
//   32-elem rows (pi_code=3): padded LDS stride 40
// D# layout per CDNA5 ISA.
// ---------------------------------------------------------------------------
#if HAVE_TDM
static __device__ __forceinline__ void tdm_load_bf16_2d(const void* gptr,
                                                        unsigned lds_off,
                                                        unsigned tile_w,
                                                        unsigned tile_h,
                                                        unsigned row_stride,
                                                        unsigned pi_code) {
    const unsigned long long ga = (unsigned long long)(uintptr_t)gptr;
    u32x4 g0;
    g0[0] = 1u;                                   // count=1 (valid user D#)
    g0[1] = lds_off;                              // lds_addr
    g0[2] = (unsigned)(ga & 0xFFFFFFFFu);         // global_addr[31:0]
    g0[3] = (unsigned)((ga >> 32) & 0x1FFFFFFu)   // global_addr[56:32]
            | (2u << 30);                         // type=2 ("image")
    i32x8 g1;
    g1[0] = (int)((1u << 16)                      // data_size = 2 bytes
                  | (1u << 20)                    // pad_enable
                  | (pi_code << 22)               // pad_interval
                  | (3u << 25));                  // pad_amount: 4 DWORDs
    g1[1] = (int)(tile_w << 16);                  // tensor_dim0[15:0]
    g1[2] = (int)((tile_w >> 16) | (tile_h << 16));  // dim0 hi | dim1 lo
    g1[3] = (int)((tile_h >> 16) | (tile_w << 16));  // dim1 hi | tile_dim0
    g1[4] = (int)(tile_h & 0xFFFFu);              // tile_dim1 (tile_dim2=0)
    g1[5] = (int)row_stride;                      // tensor_dim0_stride[31:0]
    g1[6] = 0;
    g1[7] = 0;
    const i32x4 z4 = {0, 0, 0, 0};
#if __clang_major__ >= 23
    const i32x8 z8 = {0, 0, 0, 0, 0, 0, 0, 0};
    __builtin_amdgcn_tensor_load_to_lds(g0, g1, z4, z4, z8, 0);
#else
    __builtin_amdgcn_tensor_load_to_lds(g0, g1, z4, z4, 0);
#endif
}
#define LDSOFF(p) ((unsigned)(uintptr_t)(const void*)(p))
#endif

// ---------------------------------------------------------------------------
// GEMM: C[M,N] = A[M,K] @ W[K,N] + bias[N].  128x128 block, BK=32, 4 waves.
// Wave w: rows [32w, 32w+32) x 128 cols = 16 WMMA accumulators per K-step.
// W tile staged transposed so B fragments are K-contiguous.  bf16 A tiles
// are staged by the TDM (out-projection); f32 A tiles by VALU convert.
// VT_OUT: scatter output as per-head V^T layout [B,H,hd,S] for attention.
// ---------------------------------------------------------------------------
template <typename AT, typename OT, bool VT_OUT>
__global__ __launch_bounds__(128) void gemm_wmma_kernel(
        const AT* __restrict__ A, const float* __restrict__ W,
        const float* __restrict__ bias, OT* __restrict__ C,
        int M, int N, int K) {
    constexpr int BM = 128, BN = 128, BK = 32;
    constexpr int LDA = BK + 8;   // padded LDS stride (40)
    constexpr bool A_IS_BF16 = (sizeof(AT) == 2);
    __shared__ bhalf sA[BM * LDA];
    __shared__ bhalf sBt[BN * LDA];

    const int tid  = threadIdx.x;
    const int lane = tid & 31;
    const int wave = tid >> 5;
    const int m0   = blockIdx.y * BM;
    const int n0   = blockIdx.x * BN;

    v8f acc[2][8] = {};

    for (int k0 = 0; k0 < K; k0 += BK) {
        __syncthreads();
#if HAVE_TDM
        if (A_IS_BF16) {
            if (wave == 0)
                tdm_load_bf16_2d(A + (size_t)m0 * K + k0, LDSOFF(sA),
                                 BK, BM, K, /*pi=*/3);
        } else {
            stage_tile(A + (size_t)m0 * K + k0, K, sA, LDA, BM, BK, tid, 128);
        }
#else
        stage_tile(A + (size_t)m0 * K + k0, K, sA, LDA, BM, BK, tid, 128);
#endif
        stage_w_T(W + (size_t)k0 * N + n0, N, sBt, LDA, BK, BN, tid, 128);
        if (k0 + BK < K) {  // global_prefetch_b8
            __builtin_prefetch(A + (size_t)m0 * K + k0 + BK, 0, 1);
            __builtin_prefetch(W + (size_t)(k0 + BK) * N + n0, 0, 1);
        }
#if HAVE_TDM
        if (A_IS_BF16 && wave == 0) __builtin_amdgcn_s_wait_tensorcnt(0);
#endif
        __syncthreads();

        const v16bf a0 = lds_frag_a(sA, LDA, wave * 32, 0, lane);
        const v16bf a1 = lds_frag_a(sA, LDA, wave * 32 + 16, 0, lane);
#pragma unroll
        for (int nt = 0; nt < 8; ++nt) {
            const v16bf b = lds_frag_bT(sBt, LDA, 0, nt * 16, lane);
            acc[0][nt] = wmma_bf16(a0, b, acc[0][nt]);
            acc[1][nt] = wmma_bf16(a1, b, acc[1][nt]);
        }
    }

#pragma unroll
    for (int mt = 0; mt < 2; ++mt) {
#pragma unroll
        for (int nt = 0; nt < 8; ++nt) {
            const int col = n0 + nt * 16 + (lane & 15);
            const float bv = bias[col];
#pragma unroll
            for (int j = 0; j < 8; ++j) {
                const int row = m0 + wave * 32 + mt * 16 + j + ((lane >> 4) << 3);
                const float v = acc[mt][nt][j] + bv;
                if (VT_OUT) {
                    const int b = row >> 11, s = row & (SEQ - 1);   // SEQ=2048
                    const int h = col >> 6, d = col & (HDIM - 1);
                    store_out(C + (((size_t)b * NHEADS + h) * HDIM + d) * SEQ + s, v);
                } else {
                    store_out(C + (size_t)row * N + col, v);
                }
            }
        }
    }
}

// ---------------------------------------------------------------------------
// Flash attention: grid (S/64, H, B), 128 threads (4 waves); wave owns 16
// query rows.  K tiles [key][dim], V^T tiles [dim][key] streamed through LDS
// by the TDM with double buffering (fallback: vectorized copies).
// ---------------------------------------------------------------------------
__global__ __launch_bounds__(128) void flash_attn_kernel(
        const bhalf* __restrict__ Q, const bhalf* __restrict__ Kmat,
        const bhalf* __restrict__ Vt, bhalf* __restrict__ O) {
    constexpr int LDT = HDIM + 8;   // 72, matches TDM pad descriptor
    __shared__ bhalf sQ[64 * LDT];
    __shared__ bhalf sK[2][64 * LDT];
    __shared__ bhalf sV[2][64 * LDT];
    __shared__ bhalf sP[4][16 * LDT];   // per-wave P staging

    const int tid  = threadIdx.x;
    const int lane = tid & 31;
    const int wave = tid >> 5;
    const int q0   = blockIdx.x * 64;
    const int h    = blockIdx.y;
    const int b    = blockIdx.z;
    const size_t base   = (size_t)b * SEQ * EMBED + (size_t)h * HDIM;
    const size_t vtbase = ((size_t)b * NHEADS + h) * HDIM * SEQ;

#if HAVE_TDM
    if (wave == 0) {
        tdm_load_bf16_2d(Q + base + (size_t)q0 * EMBED, LDSOFF(sQ), 64, 64, EMBED, 4);
        tdm_load_bf16_2d(Kmat + base, LDSOFF(sK[0]), 64, 64, EMBED, 4);
        tdm_load_bf16_2d(Vt + vtbase, LDSOFF(sV[0]), 64, 64, SEQ, 4);
        __builtin_amdgcn_s_wait_tensorcnt(0);
    }
    __syncthreads();
#else
    stage_tile(Q + base + (size_t)q0 * EMBED, EMBED, sQ, LDT, 64, 64, tid, 128);
    stage_tile(Kmat + base, EMBED, sK[0], LDT, 64, 64, tid, 128);
    stage_tile(Vt + vtbase, SEQ, sV[0], LDT, 64, 64, tid, 128);
    __syncthreads();
#endif

    v16bf qf[2];
    qf[0] = lds_frag_a(sQ, LDT, wave * 16, 0, lane);
    qf[1] = lds_frag_a(sQ, LDT, wave * 16, 32, lane);

    float mrun[8], lrun[8];
    v8f o[4] = {};
#pragma unroll
    for (int j = 0; j < 8; ++j) { mrun[j] = -1e30f; lrun[j] = 0.0f; }

    const float sc = 0.125f;  // 1/sqrt(64)
    int buf = 0;

    for (int kt = 0; kt < SEQ; kt += 64) {
        const int nxt = buf ^ 1;
        // prefetch next K/V tiles (DMA overlaps with this tile's compute)
        if (kt + 64 < SEQ) {
#if HAVE_TDM
            if (wave == 0) {
                tdm_load_bf16_2d(Kmat + base + (size_t)(kt + 64) * EMBED,
                                 LDSOFF(sK[nxt]), 64, 64, EMBED, 4);
                tdm_load_bf16_2d(Vt + vtbase + (kt + 64),
                                 LDSOFF(sV[nxt]), 64, 64, SEQ, 4);
            }
#else
            stage_tile(Kmat + base + (size_t)(kt + 64) * EMBED, EMBED,
                       sK[nxt], LDT, 64, 64, tid, 128);
            stage_tile(Vt + vtbase + (kt + 64), SEQ,
                       sV[nxt], LDT, 64, 64, tid, 128);
#endif
        }

        // scores S = Q (16x64) x K^T (64x64), f32 accum
        v8f s[4] = {};
#pragma unroll
        for (int d0 = 0; d0 < HDIM; d0 += 32) {
            const v16bf a = qf[d0 >> 5];
#pragma unroll
            for (int nt = 0; nt < 4; ++nt) {
                const v16bf bf = lds_frag_bT(sK[buf], LDT, d0, nt * 16, lane);
                s[nt] = wmma_bf16(a, bf, s[nt]);
            }
        }

        // online softmax; row j of a 16x16 D-tile lives in one 16-lane group
        float p[4][8], alpha[8];
#pragma unroll
        for (int j = 0; j < 8; ++j) {
            float m = -1e30f;
#pragma unroll
            for (int nt = 0; nt < 4; ++nt) { s[nt][j] *= sc; m = fmaxf(m, s[nt][j]); }
#pragma unroll
            for (int off = 8; off >= 1; off >>= 1)
                m = fmaxf(m, __shfl_xor(m, off, 32));
            const float mnew = fmaxf(mrun[j], m);
            alpha[j] = __expf(mrun[j] - mnew);
            mrun[j]  = mnew;
            float rs = 0.0f;
#pragma unroll
            for (int nt = 0; nt < 4; ++nt) {
                p[nt][j] = __expf(s[nt][j] - mnew);
                rs += p[nt][j];
            }
#pragma unroll
            for (int off = 8; off >= 1; off >>= 1)
                rs += __shfl_xor(rs, off, 32);
            lrun[j] = lrun[j] * alpha[j] + rs;
        }

        // rescale O, stash P into per-wave LDS (C-layout -> A-layout staging)
#pragma unroll
        for (int nt = 0; nt < 4; ++nt) {
#pragma unroll
            for (int j = 0; j < 8; ++j) {
                o[nt][j] *= alpha[j];
                sP[wave][(j + ((lane >> 4) << 3)) * LDT + nt * 16 + (lane & 15)] =
                    f2bf(p[nt][j]);
            }
        }

        // O += P (16x64) x V (64x64); V^T tile is K(key)-contiguous
#pragma unroll
        for (int k0 = 0; k0 < 64; k0 += 32) {
            const v16bf a = lds_frag_a(&sP[wave][0], LDT, 0, k0, lane);
#pragma unroll
            for (int nt = 0; nt < 4; ++nt) {
                const v16bf bf = lds_frag_bT(sV[buf], LDT, k0, nt * 16, lane);
                o[nt] = wmma_bf16(a, bf, o[nt]);
            }
        }

#if HAVE_TDM
        if (wave == 0) __builtin_amdgcn_s_wait_tensorcnt(0);
#endif
        __syncthreads();
        buf = nxt;
    }

    // normalize and store to concat layout [B,S,D] (bf16)
#pragma unroll
    for (int j = 0; j < 8; ++j) {
        const float inv = 1.0f / lrun[j];
        const int row = q0 + wave * 16 + j + ((lane >> 4) << 3);
#pragma unroll
        for (int nt = 0; nt < 4; ++nt) {
            O[base + (size_t)row * EMBED + nt * 16 + (lane & 15)] =
                f2bf(o[nt][j] * inv);
        }
    }
}

// ---------------------------------------------------------------------------
// Host launcher
// ---------------------------------------------------------------------------
extern "C" void kernel_launch(void* const* d_in, const int* in_sizes, int n_in,
                              void* d_out, int out_size, void* d_ws, size_t ws_size,
                              hipStream_t stream) {
    const float* x_q = (const float*)d_in[0];
    const float* x_k = (const float*)d_in[1];
    const float* x_v = (const float*)d_in[2];
    const float* wq  = (const float*)d_in[3];
    const float* bq  = (const float*)d_in[4];
    const float* wk  = (const float*)d_in[5];
    const float* bk  = (const float*)d_in[6];
    const float* wv  = (const float*)d_in[7];
    const float* bv  = (const float*)d_in[8];
    const float* wo  = (const float*)d_in[9];
    const float* bo  = (const float*)d_in[10];
    float* out = (float*)d_out;

    const int M = BATCH * SEQ;               // 4096 rows
    const size_t plane = (size_t)M * EMBED;  // elements per bf16 buffer
    bhalf* Qw  = (bhalf*)d_ws;
    bhalf* Kw  = Qw + plane;
    bhalf* Vtw = Kw + plane;                 // per-head transposed V [B,H,hd,S]
    bhalf* Aw  = Vtw + plane;                // total 32 MB of d_ws

    dim3 gg(EMBED / 128, M / 128);           // 8 x 32 blocks
    gemm_wmma_kernel<float, bhalf, false><<<gg, 128, 0, stream>>>(x_q, wq, bq, Qw, M, EMBED, EMBED);
    gemm_wmma_kernel<float, bhalf, false><<<gg, 128, 0, stream>>>(x_k, wk, bk, Kw, M, EMBED, EMBED);
    gemm_wmma_kernel<float, bhalf, true ><<<gg, 128, 0, stream>>>(x_v, wv, bv, Vtw, M, EMBED, EMBED);

    flash_attn_kernel<<<dim3(SEQ / 64, NHEADS, BATCH), 128, 0, stream>>>(Qw, Kw, Vtw, Aw);

    gemm_wmma_kernel<bhalf, float, false><<<gg, 128, 0, stream>>>(Aw, wo, bo, out, M, EMBED, EMBED);
}
